// BiLSTMTagger_18803366821909
// MI455X (gfx1250) — compile-verified, matrix operand order checked by reading, not run on hardware
//
#include <hip/hip_runtime.h>

// ---------------------------------------------------------------------------
// BiLSTM tagger for MI455X (gfx1250, wave32, WMMA 16x16x32 f16).
//  - All matmuls on v_wmma_f32_16x16x32_f16 (f16 in, f32 accumulate).
//  - GEMM: workgroup tile 32(M) x 512(N), 8 waves; each wave 32x64 with two
//    A fragments sharing four B fragments. B pre-packed so fragment loads are
//    base + constant-immediate.
//  - Recurrent scans: persistent kernel, 16 batch rows / workgroup, h in LDS
//    as f16; each wave owns a 32-wide hidden slice for all 4 gates
//    (8 accumulators) -> wave-local LSTM cell update, masked + reversible.
//    Weight loads carry an opaque per-step offset so LICM cannot hoist the
//    (t-invariant) fragment loads and spill them to scratch; they re-issue
//    each step as global_load_b128 clauses that hit the L2-resident weights.
// ---------------------------------------------------------------------------

typedef __attribute__((ext_vector_type(16))) _Float16 v16h;
typedef __attribute__((ext_vector_type(8)))  float    v8f;

#define TDIM   256
#define BATCH  64
#define EDIM   256
#define HDIM   512
#define G4H    2048          // 4*H
#define X2H    1024          // 2*H
#define NTAGS  50
#define NTAGSP 64
#define BT     (BATCH * TDIM)

__device__ __forceinline__ float sigmoidf_(float x) {
    return 1.0f / (1.0f + __expf(-x));
}

// --------------------------- zero fill (f16) -------------------------------
__global__ void bl_zero_f16(_Float16* __restrict__ p, long n) {
    long id = (long)blockIdx.x * blockDim.x + threadIdx.x;
    if (id < n) p[id] = (_Float16)0.0f;
}

// --------------------- embedding gather f32 -> f16 -------------------------
__global__ void bl_embed_gather(const int* __restrict__ x,
                                const float* __restrict__ emb,
                                _Float16* __restrict__ out) {
    int id = blockIdx.x * blockDim.x + threadIdx.x;       // over BT*EDIM
    int row = id / EDIM;
    int k   = id % EDIM;
    int tok = x[row];
    out[id] = (_Float16)emb[(size_t)tok * EDIM + k];
}

// ------------- pack weight [Nact,K] f32 row-major -> WMMA-B f16 ------------
// Fragment (nt,kb) = 16(N) x 32(K) B tile, 512 f16, at offset (nt*nkb+kb)*512;
// lane l supplies 16 contiguous f16:  n = nt*16 + (l&15),
//                                     k = kb*32 + (l>>4)*16 + e
__global__ void bl_pack_b(const float* __restrict__ W, _Float16* __restrict__ out,
                          int Nact, int Npad, int K) {
    int id = blockIdx.x * blockDim.x + threadIdx.x;       // over Npad*K
    if (id >= Npad * K) return;
    int nkb  = K >> 5;
    int e    = id & 15;
    int l    = (id >> 4) & 31;
    int frag = id >> 9;
    int kb   = frag % nkb;
    int nt   = frag / nkb;
    int n    = nt * 16 + (l & 15);
    int k    = kb * 32 + (l >> 4) * 16 + e;
    _Float16 v = (_Float16)0.0f;
    if (n < Nact) v = (_Float16)W[(size_t)n * K + k];
    out[id] = v;
}

// --------------------------- WMMA GEMM + bias ------------------------------
// C[M,Nout] = A[M,K](f16) * Bpacked + bias.  Templated on K so every B/LDS
// offset folds to a constant immediate off one per-lane base pointer.
template<int K>
__global__ __launch_bounds__(256)
void bl_wmma_gemm(const _Float16* __restrict__ A,
                  const _Float16* __restrict__ Bpk,
                  const float* __restrict__ bias,
                  float* __restrict__ Cf32,
                  _Float16* __restrict__ Cf16,
                  int Npad, int Nout) {
    constexpr int NKB = K / 32;
    __shared__ _Float16 Atile[32 * 32];
    const int tid   = threadIdx.x;
    const int wave  = tid >> 5;
    const int lane  = tid & 31;
    const int mt    = blockIdx.y;                  // 32-row M tile
    const int nbase = blockIdx.x * 512 + wave * 64;
    const bool active = (nbase < Npad);

    v8f zero = {};
    v8f acc[2][4];
#pragma unroll
    for (int h = 0; h < 2; ++h)
#pragma unroll
        for (int v = 0; v < 4; ++v) acc[h][v] = zero;

    // cooperative A staging: 32x32 f16 tile, 4 f16 (8B) per thread
    const int arow = tid >> 3;                     // 0..31
    const int acol = (tid & 7) * 4;                // 0..28
    const _Float16* Aptr = A + (size_t)(mt * 32 + arow) * K + acol;

    // per-lane bases: all further offsets are compile-time constants
    const _Float16* Bbase = Bpk + ((size_t)(nbase >> 4) * NKB * 32 + lane) * 16;
    const _Float16* a0p = &Atile[((lane & 15) +  0) * 32 + (lane >> 4) * 16];
    const _Float16* a1p = &Atile[((lane & 15) + 16) * 32 + (lane >> 4) * 16];

    for (int kb = 0; kb < NKB; ++kb) {
        *(uint2*)&Atile[arow * 32 + acol] = *(const uint2*)(Aptr + kb * 32);
        __syncthreads();
        if (active) {
            const v16h a0 = *(const v16h*)a0p;
            const v16h a1 = *(const v16h*)a1p;
            v16h b[4];
#pragma unroll
            for (int v = 0; v < 4; ++v)
                b[v] = *(const v16h*)(Bbase + (size_t)kb * 512
                                            + (size_t)v * (NKB * 512));
#pragma unroll
            for (int v = 0; v < 4; ++v) {
                acc[0][v] = __builtin_amdgcn_wmma_f32_16x16x32_f16(
                    false, a0, false, b[v], (short)0, acc[0][v], false, false);
                acc[1][v] = __builtin_amdgcn_wmma_f32_16x16x32_f16(
                    false, a1, false, b[v], (short)0, acc[1][v], false, false);
            }
            if (kb + 1 < NKB)   // global_prefetch_b8 for next K-block of B
                __builtin_prefetch((const void*)(Bbase + (size_t)(kb + 1) * 512),
                                   0, 1);
        }
        __syncthreads();
    }
    if (!active) return;

#pragma unroll
    for (int h = 0; h < 2; ++h) {
        const int mrow = mt * 32 + h * 16 + (lane >> 4) * 8;
#pragma unroll
        for (int v = 0; v < 4; ++v) {
            const int n = nbase + v * 16 + (lane & 15);
            if (n >= Nout) continue;
            const float bn = bias ? bias[n] : 0.0f;
#pragma unroll
            for (int r = 0; r < 8; ++r) {
                const size_t m = (size_t)(mrow + r);
                const float val = acc[h][v][r] + bn;
                if (Cf32) Cf32[m * Nout + n] = val;
                if (Cf16) Cf16[m * Nout + n] = (_Float16)val;
            }
        }
    }
}

// ------------------------- masked BiLSTM scan ------------------------------
// grid = (B/16, 2 directions); block = 512 thr = 16 waves.
// Wave w owns hidden slice j in [w*32, w*32+32) and computes i,f,g,o for it.
__global__ __launch_bounds__(512)
void bl_lstm_scan(const _Float16* __restrict__ preF,   // [BT, 4H] f16
                  const _Float16* __restrict__ preB,
                  const _Float16* __restrict__ whhF,   // packed B frags (K=H)
                  const _Float16* __restrict__ whhB,
                  const int* __restrict__ lengths,
                  _Float16* __restrict__ xout) {       // [BT, 2H] (pre-zeroed)
    constexpr int NKB = HDIM / 32;                     // 16
    __shared__ _Float16 hbuf[16 * HDIM];               // h state, A-readable
    const int tid  = threadIdx.x;
    const int wave = tid >> 5;                         // 0..15
    const int lane = tid & 31;
    const int hi   = lane >> 4;                        // 0/1
    const int blk  = blockIdx.x;                       // batch tile
    const int dir  = blockIdx.y;                       // 0 fwd, 1 bwd
    const _Float16* pre = dir ? preB : preF;
    const _Float16* whh = dir ? whhB : whhF;
    const int col_off = dir ? HDIM : 0;

    for (int i = tid; i < 16 * HDIM; i += 512) hbuf[i] = (_Float16)0.0f;

    int lenv[8];
#pragma unroll
    for (int r = 0; r < 8; ++r)
        lenv[r] = lengths[blk * 16 + hi * 8 + r];

    float cst[2][8];
#pragma unroll
    for (int s = 0; s < 2; ++s)
#pragma unroll
        for (int r = 0; r < 8; ++r) cst[s][r] = 0.0f;

    // per-lane bases: fragment nt = g*32 + wave*2 + s  ->
    //   offset (g*32+s)*NKB*512 + kb*512 from Bbase (constant immediates)
    const _Float16* Bbase = whh + ((size_t)(wave * 2) * NKB * 32 + lane) * 16;
    const _Float16* hrow  = &hbuf[(lane & 15) * HDIM + hi * 16];

    v8f zero = {};
    __syncthreads();

    int tdep = 0;   // opaque 0: defeats LICM hoist+spill of weight loads
    for (int t = 0; t < TDIM; ++t) {
        asm volatile("" : "+s"(tdep));
        const _Float16* Bt = Bbase + tdep;

        // ---- gates(partial) = h @ Whh^T for this wave's 32 cols x 4 gates
        v8f acc[8];
#pragma unroll
        for (int u = 0; u < 8; ++u) acc[u] = zero;
#pragma unroll
        for (int kb = 0; kb < NKB; ++kb) {
            const v16h a = *(const v16h*)(hrow + kb * 32);
#pragma unroll
            for (int half = 0; half < 2; ++half) {     // 2 batches of 4 frags
                v16h b[4];
#pragma unroll
                for (int q = 0; q < 4; ++q) {
                    const int g = half * 2 + (q >> 1); // gate 0..3
                    const int s = q & 1;
                    b[q] = *(const v16h*)(Bt
                        + (size_t)kb * 512
                        + (size_t)(g * 32 + s) * (NKB * 512));
                }
#pragma unroll
                for (int q = 0; q < 4; ++q) {
                    const int u = half * 4 + q;        // == g*2+s
                    acc[u] = __builtin_amdgcn_wmma_f32_16x16x32_f16(
                        false, a, false, b[q], (short)0, acc[u], false, false);
                }
            }
        }
        __syncthreads();   // all hbuf reads for step t done

        // ---- LSTM cell, wave-local (masked, reversal folded into tt)
#pragma unroll
        for (int s = 0; s < 2; ++s) {
#pragma unroll
            for (int r = 0; r < 8; ++r) {
                const int row = hi * 8 + r;
                const int b   = blk * 16 + row;
                const int len = lenv[r];
                const bool valid = (t < len);
                const int tt = dir ? (len - 1 - t) : t;
                const int j  = wave * 32 + s * 16 + (lane & 15);
                float pi = 0.f, pf = 0.f, pg = 0.f, po = 0.f;
                if (valid) {
                    const _Float16* pb = pre + ((size_t)b * TDIM + tt) * G4H + j;
                    pi = (float)pb[0 * HDIM];
                    pf = (float)pb[1 * HDIM];
                    pg = (float)pb[2 * HDIM];
                    po = (float)pb[3 * HDIM];
                }
                const float iv = sigmoidf_(acc[0 * 2 + s][r] + pi);
                const float fv = sigmoidf_(acc[1 * 2 + s][r] + pf);
                const float gv = tanhf   (acc[2 * 2 + s][r] + pg);
                const float ov = sigmoidf_(acc[3 * 2 + s][r] + po);
                const float cn = fv * cst[s][r] + iv * gv;
                const float hn = ov * tanhf(cn);
                if (valid) {
                    cst[s][r] = cn;
                    hbuf[row * HDIM + j] = (_Float16)hn;
                    xout[((size_t)b * TDIM + tt) * X2H + col_off + j] = (_Float16)hn;
                }
            }
        }
        __syncthreads();   // h writes visible before next step's WMMA reads
    }
}

// ---------------------------------------------------------------------------
extern "C" void kernel_launch(void* const* d_in, const int* in_sizes, int n_in,
                              void* d_out, int out_size, void* d_ws, size_t ws_size,
                              hipStream_t stream) {
    const int*   x        = (const int*)  d_in[0];
    const int*   lengths  = (const int*)  d_in[1];
    const float* emb      = (const float*)d_in[2];
    const float* w_ih_f1  = (const float*)d_in[3];
    const float* w_hh_f1  = (const float*)d_in[4];
    const float* b_f1     = (const float*)d_in[5];
    const float* w_ih_b1  = (const float*)d_in[6];
    const float* w_hh_b1  = (const float*)d_in[7];
    const float* b_b1     = (const float*)d_in[8];
    const float* w_ih_f2  = (const float*)d_in[9];
    const float* w_hh_f2  = (const float*)d_in[10];
    const float* b_f2     = (const float*)d_in[11];
    const float* w_ih_b2  = (const float*)d_in[12];
    const float* w_hh_b2  = (const float*)d_in[13];
    const float* b_b2     = (const float*)d_in[14];
    const float* cls_w    = (const float*)d_in[15];
    const float* cls_b    = (const float*)d_in[16];
    float* logits = (float*)d_out;

    // ---- workspace carve-out (256B aligned)
    char* ws = (char*)d_ws;
    size_t off = 0;
    auto carve = [&](size_t bytes) {
        void* p = ws + off;
        off = (off + bytes + 255) & ~(size_t)255;
        return p;
    };
    _Float16* e16   = (_Float16*)carve((size_t)BT * EDIM * 2);
    _Float16* preF  = (_Float16*)carve((size_t)BT * G4H * 2);
    _Float16* preB  = (_Float16*)carve((size_t)BT * G4H * 2);
    _Float16* x1    = (_Float16*)carve((size_t)BT * X2H * 2);
    _Float16* x2    = (_Float16*)carve((size_t)BT * X2H * 2);
    _Float16* pk_ih_f1 = (_Float16*)carve((size_t)G4H * EDIM * 2);
    _Float16* pk_ih_b1 = (_Float16*)carve((size_t)G4H * EDIM * 2);
    _Float16* pk_hh_f1 = (_Float16*)carve((size_t)G4H * HDIM * 2);
    _Float16* pk_hh_b1 = (_Float16*)carve((size_t)G4H * HDIM * 2);
    _Float16* pk_ih_f2 = (_Float16*)carve((size_t)G4H * X2H * 2);
    _Float16* pk_ih_b2 = (_Float16*)carve((size_t)G4H * X2H * 2);
    _Float16* pk_hh_f2 = (_Float16*)carve((size_t)G4H * HDIM * 2);
    _Float16* pk_hh_b2 = (_Float16*)carve((size_t)G4H * HDIM * 2);
    _Float16* pk_cls   = (_Float16*)carve((size_t)NTAGSP * X2H * 2);

    // ---- pack all weights to WMMA-B f16 fragment layout
    auto pack = [&](const float* W, _Float16* dst, int Nact, int Npad, int K) {
        int n = Npad * K;
        bl_pack_b<<<(n + 255) / 256, 256, 0, stream>>>(W, dst, Nact, Npad, K);
    };
    pack(w_ih_f1, pk_ih_f1, G4H, G4H, EDIM);
    pack(w_ih_b1, pk_ih_b1, G4H, G4H, EDIM);
    pack(w_hh_f1, pk_hh_f1, G4H, G4H, HDIM);
    pack(w_hh_b1, pk_hh_b1, G4H, G4H, HDIM);
    pack(w_ih_f2, pk_ih_f2, G4H, G4H, X2H);
    pack(w_ih_b2, pk_ih_b2, G4H, G4H, X2H);
    pack(w_hh_f2, pk_hh_f2, G4H, G4H, HDIM);
    pack(w_hh_b2, pk_hh_b2, G4H, G4H, HDIM);
    pack(cls_w,   pk_cls,   NTAGS, NTAGSP, X2H);

    // ---- zero concat buffers (padding positions must read as 0)
    {
        long n = (long)BT * X2H;
        bl_zero_f16<<<(unsigned)((n + 255) / 256), 256, 0, stream>>>(x1, n);
        bl_zero_f16<<<(unsigned)((n + 255) / 256), 256, 0, stream>>>(x2, n);
    }

    // ---- embedding gather
    bl_embed_gather<<<(BT * EDIM) / 256, 256, 0, stream>>>(x, emb, e16);

    // ---- layer 1: input projections then bidirectional scan
    dim3 gemm_blk(256);
    dim3 g1((G4H + 511) / 512, BT / 32);
    bl_wmma_gemm<EDIM><<<g1, gemm_blk, 0, stream>>>(e16, pk_ih_f1, b_f1,
                                                    nullptr, preF, G4H, G4H);
    bl_wmma_gemm<EDIM><<<g1, gemm_blk, 0, stream>>>(e16, pk_ih_b1, b_b1,
                                                    nullptr, preB, G4H, G4H);
    bl_lstm_scan<<<dim3(BATCH / 16, 2), 512, 0, stream>>>(
        preF, preB, pk_hh_f1, pk_hh_b1, lengths, x1);

    // ---- layer 2 (reuse pre buffers)
    bl_wmma_gemm<X2H><<<g1, gemm_blk, 0, stream>>>(x1, pk_ih_f2, b_f2,
                                                   nullptr, preF, G4H, G4H);
    bl_wmma_gemm<X2H><<<g1, gemm_blk, 0, stream>>>(x1, pk_ih_b2, b_b2,
                                                   nullptr, preB, G4H, G4H);
    bl_lstm_scan<<<dim3(BATCH / 16, 2), 512, 0, stream>>>(
        preF, preB, pk_hh_f2, pk_hh_b2, lengths, x2);

    // ---- classifier: [BT,1024] x [1024,64pad] -> f32 logits [BT,50]
    dim3 gc((NTAGSP + 511) / 512, BT / 32);
    bl_wmma_gemm<X2H><<<gc, gemm_blk, 0, stream>>>(x2, pk_cls, cls_b,
                                                   logits, nullptr, NTAGSP, NTAGS);

    (void)in_sizes; (void)n_in; (void)out_size; (void)ws_size;
}
